// GRU_Decoder_10608569221466
// MI455X (gfx1250) — matrix operand order repaired and checked by reading, not compiled, not run on hardware
//
#include <hip/hip_runtime.h>

// ---------------------------------------------------------------------------
// GRU decoder for MI455X (gfx1250): bf16 WMMA matmuls, fp32 accumulation/state.
// B=128, T=128, H=E=1024.  X@W_ih0^T hoisted out of the time loop as one big
// parallel GEMM; recurrent steps are 3 fused WMMA kernels per step.
// LDS tiles staged via the Tensor Data Mover when the builtin is available
// (one tensor_load_to_lds per tile + s_wait_tensorcnt), else via register-
// batched global->LDS copies.
// ---------------------------------------------------------------------------

typedef __bf16 bf16;
typedef bf16  v16bf __attribute__((ext_vector_type(16)));
typedef bf16  v8bf  __attribute__((ext_vector_type(8)));
typedef float v8f   __attribute__((ext_vector_type(8)));

#define WMMA_BF16(acc, a, b)                                                   \
  acc = __builtin_amdgcn_wmma_f32_16x16x32_bf16(false, (a), false, (b),        \
                                                (short)0, (acc), false, false)

// ------------------------- TDM staging support ------------------------------
#if defined(__has_builtin)
#if __has_builtin(__builtin_amdgcn_tensor_load_to_lds) &&                      \
    __has_builtin(__builtin_amdgcn_s_wait_tensorcnt)
#define CDNA5_TDM 1
#endif
#endif
#ifndef CDNA5_TDM
#define CDNA5_TDM 0
#endif

#if CDNA5_TDM
typedef unsigned int v4ui __attribute__((ext_vector_type(4)));
typedef int v8si __attribute__((ext_vector_type(8)));
typedef int v4si __attribute__((ext_vector_type(4)));

// Generic LDS pointer -> 32-bit LDS byte address (low dword of aperture addr).
__device__ __forceinline__ unsigned lds_addr_of(const void* p) {
  return (unsigned)(unsigned long long)p;
}

// Issue one TDM 2-D tile load: rows x 64 bf16 tile from a row-major tensor
// with leading dimension src_ld_elems, landing compact (stride 64) in LDS.
// D# layout per CDNA5 ISA 8.3/8.4 (group0: count|lds|global|type=2;
// group1: data_size=2B, tensor dims, tile dims, dim0 stride).
__device__ __forceinline__ void tdm_load_tile64(unsigned lds_addr,
                                                const bf16* gsrc,
                                                unsigned src_ld_elems,
                                                unsigned rows) {
  unsigned long long ga = (unsigned long long)(const void*)gsrc;
  v4ui g0;
  g0[0] = 1u;                                   // count=1, user descriptor
  g0[1] = lds_addr;                             // LDS byte address
  g0[2] = (unsigned)(ga & 0xffffffffull);       // global_addr[31:0]
  g0[3] = (unsigned)((ga >> 32) & 0x01ffffffull) | (2u << 30);  // [56:32]|type
  v8si g1;
  g1[0] = (int)0x00010000u;                               // data_size=1 (2B)
  g1[1] = (int)((src_ld_elems & 0xffffu) << 16);          // tensor_dim0 lo16
  g1[2] = (int)(((src_ld_elems >> 16) & 0xffffu) |
                ((rows & 0xffffu) << 16));                // dim0 hi | dim1 lo
  g1[3] = (int)(((rows >> 16) & 0xffffu) | (64u << 16));  // dim1 hi | tile_dim0
  g1[4] = (int)(rows & 0xffffu);                          // tile_dim1 (dim2=0)
  g1[5] = (int)src_ld_elems;                              // dim0_stride lo32
  g1[6] = 0;
  g1[7] = 0;
  v4si z4 = {};
#if defined(__clang_major__) && (__clang_major__ >= 23)
  v8si z8 = {};
  __builtin_amdgcn_tensor_load_to_lds(g0, g1, z4, z4, z8, 0);
#else
  __builtin_amdgcn_tensor_load_to_lds(g0, g1, z4, z4, 0);
#endif
}
#endif  // CDNA5_TDM

__device__ __forceinline__ float sigmoidf_(float x) {
  return 1.0f / (1.0f + __expf(-x));
}

// A fragment (16x32 bf16, MxK). Per ISA: lanes 0-15 hold M=lane, K={0..7,16..23};
// lanes 16-31 hold M=lane-16, K={8..15,24..31}.  smem row-major [m][ld].
__device__ __forceinline__ v16bf frag_a(const bf16* smem, int ld, int m0,
                                        int kbase, int lane) {
  const bf16* p = smem + (size_t)(m0 + (lane & 15)) * ld + kbase +
                  ((lane >> 4) << 3);
  v8bf lo = *(const v8bf*)p;
  v8bf hi = *(const v8bf*)(p + 16);
  v16bf a;
#pragma unroll
  for (int i = 0; i < 8; ++i) { a[i] = lo[i]; a[i + 8] = hi[i]; }
  return a;
}

// B fragment (32x16 bf16, KxN). lanes 0-15: N=lane, K={0..7,16..23}; lanes 16-31:
// N=lane-16, K={8..15,24..31}.  smem holds weight tile row-major [n][ld] (n = out col).
__device__ __forceinline__ v16bf frag_b(const bf16* smem, int ld, int n0,
                                        int kbase, int lane) {
  const bf16* p = smem + (size_t)(n0 + (lane & 15)) * ld + kbase +
                  ((lane >> 4) << 3);
  v8bf lo = *(const v8bf*)p;
  v8bf hi = *(const v8bf*)(p + 16);
  v16bf b;
#pragma unroll
  for (int i = 0; i < 8; ++i) { b[i] = lo[i]; b[i + 8] = hi[i]; }
  return b;
}

// Fallback staging: rows x 64 bf16 tile, register-batched so all global loads
// issue before any LDS store (partial loadcnt waits instead of full drains).
template <int ROWS>
__device__ __forceinline__ void stage_tile64(bf16* dst, const bf16* src,
                                             int src_ld, int tid) {
  constexpr int ITER = (ROWS * 8) / 256;  // 16B chunks per thread
  uint4 tmp[ITER];
#pragma unroll
  for (int i = 0; i < ITER; ++i) {
    int idx = tid + i * 256;
    int r = idx >> 3, c = (idx & 7) << 3;
    tmp[i] = *(const uint4*)(src + (size_t)r * src_ld + c);
  }
#pragma unroll
  for (int i = 0; i < ITER; ++i) {
    int idx = tid + i * 256;
    int r = idx >> 3, c = (idx & 7) << 3;
    *(uint4*)(dst + r * 64 + c) = tmp[i];
  }
}

// ---------------------------------------------------------------------------
// Cast fp32 -> bf16
__global__ void k_cast(const float* __restrict__ src, bf16* __restrict__ dst,
                       int n) {
  for (int i = blockIdx.x * blockDim.x + threadIdx.x; i < n;
       i += gridDim.x * blockDim.x)
    dst[i] = (bf16)src[i];
}

// ---------------------------------------------------------------------------
// C[M,N] = A[M,K] @ W[N,K]^T   (fp32 out).  Tile: 128M x 64N, K chunks of 64.
__global__ __launch_bounds__(256) void k_gemm_xw(const bf16* __restrict__ A,
                                                 const bf16* __restrict__ W,
                                                 float* __restrict__ C, int M,
                                                 int N, int K) {
  __shared__ bf16 As[128 * 64];
  __shared__ bf16 Bs[64 * 64];
  const int tid = threadIdx.x, lane = tid & 31, w = tid >> 5;
  const int row0 = blockIdx.y * 128, n0 = blockIdx.x * 64;
  v8f acc[4] = {};
  for (int kk = 0; kk < K; kk += 64) {
#if CDNA5_TDM
    if (tid < 32) {
      tdm_load_tile64(lds_addr_of(As), A + (size_t)row0 * K + kk, K, 128);
      tdm_load_tile64(lds_addr_of(Bs), W + (size_t)n0 * K + kk, K, 64);
      __builtin_amdgcn_s_wait_tensorcnt(0);
    }
#else
    stage_tile64<128>(As, A + (size_t)row0 * K + kk, K, tid);
    stage_tile64<64>(Bs, W + (size_t)n0 * K + kk, K, tid);
    if (kk + 64 < K) {  // hint next K-chunk into cache (global_prefetch_b8)
      __builtin_prefetch(A + (size_t)(row0 + (tid >> 1)) * K + kk + 64 +
                             ((tid & 1) << 5), 0, 1);
      __builtin_prefetch(W + (size_t)(n0 + (tid & 63)) * K + kk + 64, 0, 1);
    }
#endif
    __syncthreads();
#pragma unroll
    for (int ks = 0; ks < 64; ks += 32) {
      v16bf a = frag_a(As, 64, 16 * w, ks, lane);
#pragma unroll
      for (int s = 0; s < 4; ++s) {
        v16bf b = frag_b(Bs, 64, 16 * s, ks, lane);
        WMMA_BF16(acc[s], a, b);
      }
    }
    __syncthreads();
  }
  const int mloc = 16 * w + ((lane >> 4) << 3);
  const int ncol = n0 + (lane & 15);
#pragma unroll
  for (int s = 0; s < 4; ++s)
#pragma unroll
    for (int v = 0; v < 8; ++v)
      C[(size_t)(row0 + mloc + v) * N + ncol + 16 * s] = acc[s][v];
}

// ---------------------------------------------------------------------------
// Step kernel 1: h1 = GRU(x_t, h0) with precomputed gi = GI0[:, t, :].
// Computes gh = h0 @ W_hh0^T for the 3 gates, fuses nonlinearity.
__global__ __launch_bounds__(256) void k_gru1(
    const float* __restrict__ GI0,  // [B*T, 3H]
    const bf16* __restrict__ h0b,   // [B,H] bf16
    const bf16* __restrict__ Whh,   // [3H,H] bf16
    const float* __restrict__ h0f,  // [B,H] fp32
    bf16* __restrict__ h1b, int t) {
  const int H = 1024, T = 128;
  __shared__ bf16 As[128 * 64];
  __shared__ bf16 Bs[3][64 * 64];
  const int tid = threadIdx.x, lane = tid & 31, w = tid >> 5;
  const int n0 = blockIdx.x * 64;
  v8f aR[4] = {}, aZ[4] = {}, aN[4] = {};
  for (int kk = 0; kk < H; kk += 64) {
#if CDNA5_TDM
    if (tid < 32) {
      tdm_load_tile64(lds_addr_of(As), h0b + kk, H, 128);
      tdm_load_tile64(lds_addr_of(Bs[0]), Whh + (size_t)(0 * H + n0) * H + kk, H, 64);
      tdm_load_tile64(lds_addr_of(Bs[1]), Whh + (size_t)(1 * H + n0) * H + kk, H, 64);
      tdm_load_tile64(lds_addr_of(Bs[2]), Whh + (size_t)(2 * H + n0) * H + kk, H, 64);
      __builtin_amdgcn_s_wait_tensorcnt(0);
    }
#else
    stage_tile64<128>(As, h0b + kk, H, tid);
#pragma unroll
    for (int g = 0; g < 3; ++g)
      stage_tile64<64>(Bs[g], Whh + (size_t)(g * H + n0) * H + kk, H, tid);
#endif
    __syncthreads();
#pragma unroll
    for (int ks = 0; ks < 64; ks += 32) {
      v16bf a = frag_a(As, 64, 16 * w, ks, lane);
#pragma unroll
      for (int s = 0; s < 4; ++s) {
        v16bf bR = frag_b(Bs[0], 64, 16 * s, ks, lane);
        WMMA_BF16(aR[s], a, bR);
        v16bf bZ = frag_b(Bs[1], 64, 16 * s, ks, lane);
        WMMA_BF16(aZ[s], a, bZ);
        v16bf bN = frag_b(Bs[2], 64, 16 * s, ks, lane);
        WMMA_BF16(aN[s], a, bN);
      }
    }
    __syncthreads();
  }
  const int mloc = 16 * w + ((lane >> 4) << 3);
  const int ncol = n0 + (lane & 15);
#pragma unroll
  for (int s = 0; s < 4; ++s)
#pragma unroll
    for (int v = 0; v < 8; ++v) {
      int m = mloc + v, n = ncol + 16 * s;
      size_t gb = ((size_t)m * T + t) * (3 * H);
      float r = sigmoidf_(GI0[gb + n] + aR[s][v]);
      float z = sigmoidf_(GI0[gb + H + n] + aZ[s][v]);
      float ng = tanhf(GI0[gb + 2 * H + n] + r * aN[s][v]);
      float h1 = (1.0f - z) * ng + z * h0f[(size_t)m * H + n];
      h1b[(size_t)m * H + n] = (bf16)h1;
    }
}

// ---------------------------------------------------------------------------
// Step kernel 2: h2 = GRU(h1, h1_prev).  r/z accumulate over concatenated
// K=2048 [h1 | h1_prev]; n-gate keeps i_n and h_n separate.
__global__ __launch_bounds__(256) void k_gru2(
    const bf16* __restrict__ h1b, const bf16* __restrict__ hpb,
    const float* __restrict__ hpf, const bf16* __restrict__ Wih,
    const bf16* __restrict__ Whh, float* __restrict__ h2f,
    bf16* __restrict__ h2b) {
  const int H = 1024;
  __shared__ bf16 As[128 * 64];
  __shared__ bf16 Bs[3][32 * 64];
  const int tid = threadIdx.x, lane = tid & 31, w = tid >> 5;
  const int n0 = blockIdx.x * 32;
  v8f aR[2] = {}, aZ[2] = {}, aI[2] = {}, aH[2] = {};
  for (int kk = 0; kk < 2 * H; kk += 64) {
    const bool first = kk < H;
    const bf16* Asrc = first ? h1b : hpb;
    const bf16* Wsrc = first ? Wih : Whh;
    const int k = first ? kk : kk - H;
#if CDNA5_TDM
    if (tid < 32) {
      tdm_load_tile64(lds_addr_of(As), Asrc + k, H, 128);
      tdm_load_tile64(lds_addr_of(Bs[0]), Wsrc + (size_t)(0 * H + n0) * H + k, H, 32);
      tdm_load_tile64(lds_addr_of(Bs[1]), Wsrc + (size_t)(1 * H + n0) * H + k, H, 32);
      tdm_load_tile64(lds_addr_of(Bs[2]), Wsrc + (size_t)(2 * H + n0) * H + k, H, 32);
      __builtin_amdgcn_s_wait_tensorcnt(0);
    }
#else
    stage_tile64<128>(As, Asrc + k, H, tid);
#pragma unroll
    for (int g = 0; g < 3; ++g)
      stage_tile64<32>(Bs[g], Wsrc + (size_t)(g * H + n0) * H + k, H, tid);
#endif
    __syncthreads();
#pragma unroll
    for (int ks = 0; ks < 64; ks += 32) {
      v16bf a = frag_a(As, 64, 16 * w, ks, lane);
#pragma unroll
      for (int s = 0; s < 2; ++s) {
        v16bf bR = frag_b(Bs[0], 64, 16 * s, ks, lane);
        WMMA_BF16(aR[s], a, bR);
        v16bf bZ = frag_b(Bs[1], 64, 16 * s, ks, lane);
        WMMA_BF16(aZ[s], a, bZ);
        v16bf bN = frag_b(Bs[2], 64, 16 * s, ks, lane);
        if (first) { WMMA_BF16(aI[s], a, bN); }
        else       { WMMA_BF16(aH[s], a, bN); }
      }
    }
    __syncthreads();
  }
  const int mloc = 16 * w + ((lane >> 4) << 3);
  const int ncol = n0 + (lane & 15);
#pragma unroll
  for (int s = 0; s < 2; ++s)
#pragma unroll
    for (int v = 0; v < 8; ++v) {
      int m = mloc + v, n = ncol + 16 * s;
      float r = sigmoidf_(aR[s][v]);
      float z = sigmoidf_(aZ[s][v]);
      float ng = tanhf(aI[s][v] + r * aH[s][v]);
      float hp = hpf[(size_t)m * H + n];
      float h2 = (1.0f - z) * ng + z * hp;
      h2f[(size_t)m * H + n] = h2;
      h2b[(size_t)m * H + n] = (bf16)h2;
    }
}

// ---------------------------------------------------------------------------
// Step kernel 3: h0_new = tanh([h0 | h2] @ W_lin^T); writes output row t.
__global__ __launch_bounds__(256) void k_lin(
    const bf16* __restrict__ h0b, const bf16* __restrict__ h2b,
    const bf16* __restrict__ Wlin,  // [H, 2H]
    float* __restrict__ out, float* __restrict__ h0f_new,
    bf16* __restrict__ h0b_new, int t) {
  const int H = 1024, T = 128;
  __shared__ bf16 As[128 * 64];
  __shared__ bf16 Bs[64 * 64];
  const int tid = threadIdx.x, lane = tid & 31, w = tid >> 5;
  const int n0 = blockIdx.x * 64;
  v8f acc[4] = {};
  for (int kk = 0; kk < 2 * H; kk += 64) {
    const bf16* Asrc = (kk < H) ? h0b : h2b;
    const int k = (kk < H) ? kk : kk - H;
#if CDNA5_TDM
    if (tid < 32) {
      tdm_load_tile64(lds_addr_of(As), Asrc + k, H, 128);
      tdm_load_tile64(lds_addr_of(Bs), Wlin + (size_t)n0 * (2 * H) + kk, 2 * H, 64);
      __builtin_amdgcn_s_wait_tensorcnt(0);
    }
#else
    stage_tile64<128>(As, Asrc + k, H, tid);
    stage_tile64<64>(Bs, Wlin + (size_t)n0 * (2 * H) + kk, 2 * H, tid);
#endif
    __syncthreads();
#pragma unroll
    for (int ks = 0; ks < 64; ks += 32) {
      v16bf a = frag_a(As, 64, 16 * w, ks, lane);
#pragma unroll
      for (int s = 0; s < 4; ++s) {
        v16bf b = frag_b(Bs, 64, 16 * s, ks, lane);
        WMMA_BF16(acc[s], a, b);
      }
    }
    __syncthreads();
  }
  const int mloc = 16 * w + ((lane >> 4) << 3);
  const int ncol = n0 + (lane & 15);
#pragma unroll
  for (int s = 0; s < 4; ++s)
#pragma unroll
    for (int v = 0; v < 8; ++v) {
      int m = mloc + v, n = ncol + 16 * s;
      float y = tanhf(acc[s][v]);
      out[((size_t)m * T + t) * H + n] = y;
      h0f_new[(size_t)m * H + n] = y;
      h0b_new[(size_t)m * H + n] = (bf16)y;
    }
}

// ---------------------------------------------------------------------------
// Initialize recurrent state from G[2,B,H].
__global__ void k_init_state(const float* __restrict__ G, float* h0f,
                             bf16* h0b, float* hpf, bf16* hpb) {
  int i = blockIdx.x * blockDim.x + threadIdx.x;
  if (i < 128 * 1024) {
    float a = G[i];
    float b = G[128 * 1024 + i];
    h0f[i] = a; h0b[i] = (bf16)a;
    hpf[i] = b; hpb[i] = (bf16)b;
  }
}

// ---------------------------------------------------------------------------
// LayerNorm per row (H=1024): unbiased std, (std + eps) denominator.
__global__ __launch_bounds__(256) void k_layernorm(float* __restrict__ out,
                                                   const float* __restrict__ a2,
                                                   const float* __restrict__ b2) {
  const int H = 1024;
  __shared__ float red[2][8];
  float* row = out + (size_t)blockIdx.x * H;
  const int tid = threadIdx.x, lane = tid & 31, w = tid >> 5;
  float v[4], s = 0.f, s2 = 0.f;
#pragma unroll
  for (int i = 0; i < 4; ++i) {
    v[i] = row[tid + 256 * i];
    s += v[i];
    s2 += v[i] * v[i];
  }
#pragma unroll
  for (int off = 16; off > 0; off >>= 1) {
    s += __shfl_down(s, off, 32);
    s2 += __shfl_down(s2, off, 32);
  }
  if (lane == 0) { red[0][w] = s; red[1][w] = s2; }
  __syncthreads();
  if (w == 0) {
    float a = (lane < 8) ? red[0][lane] : 0.f;
    float b = (lane < 8) ? red[1][lane] : 0.f;
#pragma unroll
    for (int off = 4; off > 0; off >>= 1) {
      a += __shfl_down(a, off, 32);
      b += __shfl_down(b, off, 32);
    }
    if (lane == 0) { red[0][0] = a; red[1][0] = b; }
  }
  __syncthreads();
  float mean = red[0][0] * (1.0f / 1024.0f);
  float var = (red[1][0] - 1024.0f * mean * mean) * (1.0f / 1023.0f);
  float inv = 1.0f / (sqrtf(fmaxf(var, 0.0f)) + 1e-6f);
#pragma unroll
  for (int i = 0; i < 4; ++i) {
    int c = tid + 256 * i;
    row[c] = a2[c] * (v[i] - mean) * inv + b2[c];
  }
}

// ---------------------------------------------------------------------------
extern "C" void kernel_launch(void* const* d_in, const int* in_sizes, int n_in,
                              void* d_out, int out_size, void* d_ws,
                              size_t ws_size, hipStream_t stream) {
  (void)in_sizes; (void)n_in; (void)out_size; (void)ws_size;
  const int B = 128, T = 128, H = 1024, E = 1024;

  const float* X     = (const float*)d_in[0];
  const float* G     = (const float*)d_in[1];
  /* d_in[2] = L : dead code in reference */
  const float* W_ih0 = (const float*)d_in[3];
  const float* W_hh0 = (const float*)d_in[4];
  const float* W_ih1 = (const float*)d_in[5];
  const float* W_hh1 = (const float*)d_in[6];
  const float* W_lin = (const float*)d_in[7];
  const float* a2    = (const float*)d_in[8];
  const float* b2    = (const float*)d_in[9];
  float* out = (float*)d_out;

  // Workspace carve-out (256B aligned).
  char* p = (char*)d_ws;
  auto carve = [&](size_t bytes) {
    void* r = (void*)p;
    p += (bytes + 255) & ~(size_t)255;
    return r;
  };
  bf16* Xb     = (bf16*)carve((size_t)B * T * E * sizeof(bf16));
  bf16* Wih0b  = (bf16*)carve((size_t)3 * H * E * sizeof(bf16));
  bf16* Whh0b  = (bf16*)carve((size_t)3 * H * H * sizeof(bf16));
  bf16* Wih1b  = (bf16*)carve((size_t)3 * H * H * sizeof(bf16));
  bf16* Whh1b  = (bf16*)carve((size_t)3 * H * H * sizeof(bf16));
  bf16* Wlinb  = (bf16*)carve((size_t)H * 2 * H * sizeof(bf16));
  float* GI0   = (float*)carve((size_t)B * T * 3 * H * sizeof(float));
  float* h0f[2] = {(float*)carve((size_t)B * H * 4), (float*)carve((size_t)B * H * 4)};
  bf16*  h0b[2] = {(bf16*)carve((size_t)B * H * 2), (bf16*)carve((size_t)B * H * 2)};
  float* hpf[2] = {(float*)carve((size_t)B * H * 4), (float*)carve((size_t)B * H * 4)};
  bf16*  hpb[2] = {(bf16*)carve((size_t)B * H * 2), (bf16*)carve((size_t)B * H * 2)};
  bf16*  h1b    = (bf16*)carve((size_t)B * H * 2);

  // Phase 0: casts to bf16.
  k_cast<<<2048, 256, 0, stream>>>(X, Xb, B * T * E);
  k_cast<<<256, 256, 0, stream>>>(W_ih0, Wih0b, 3 * H * E);
  k_cast<<<256, 256, 0, stream>>>(W_hh0, Whh0b, 3 * H * H);
  k_cast<<<256, 256, 0, stream>>>(W_ih1, Wih1b, 3 * H * H);
  k_cast<<<256, 256, 0, stream>>>(W_hh1, Whh1b, 3 * H * H);
  k_cast<<<256, 256, 0, stream>>>(W_lin, Wlinb, H * 2 * H);

  // Phase 1: GI0 = X @ W_ih0^T  (parallel over all B*T rows).
  {
    dim3 grid((3 * H) / 64, (B * T) / 128);
    k_gemm_xw<<<grid, 256, 0, stream>>>(Xb, Wih0b, GI0, B * T, 3 * H, E);
  }

  // Phase 2: recurrent loop.
  k_init_state<<<(B * H) / 256, 256, 0, stream>>>(G, h0f[0], h0b[0], hpf[0],
                                                  hpb[0]);
  int cur = 0;
  for (int t = 0; t < T; ++t) {
    int nxt = cur ^ 1;
    k_gru1<<<H / 64, 256, 0, stream>>>(GI0, h0b[cur], Whh0b, h0f[cur], h1b, t);
    k_gru2<<<H / 32, 256, 0, stream>>>(h1b, hpb[cur], hpf[cur], Wih1b, Whh1b,
                                       hpf[nxt], hpb[nxt]);
    k_lin<<<H / 64, 256, 0, stream>>>(h0b[cur], hpb[nxt], Wlinb, out, h0f[nxt],
                                      h0b[nxt], t);
    cur = nxt;
  }

  // Phase 3: LayerNorm over last dim, in place on d_out.
  k_layernorm<<<B * T, 256, 0, stream>>>(out, a2, b2);
}
